// HypLinear_38276748542432
// MI455X (gfx1250) — matrix operand order repaired and checked by reading, not compiled, not run on hardware
//
#include <hip/hip_runtime.h>
#include <math.h>

// ---------------------------------------------------------------------------
// HypLinear for MI455X (gfx1250, wave32).
// GEMM mx = x @ W^T done with V_WMMA_F32_16X16X32_BF16 using a bf16x3
// split-precision scheme (hi*hi + hi*lo + lo*hi, fp32 accumulate).
// Epilogue kernels implement the Poincare-ball math per row.
// ---------------------------------------------------------------------------

typedef __attribute__((ext_vector_type(16))) __bf16 v16bf;
typedef __attribute__((ext_vector_type(8)))  __bf16 v8bf;
typedef __attribute__((ext_vector_type(8)))  float  v8f;

#define N_ROWS 32768
#define DIN    1024
#define DOUT   1024
#define BM     128
#define BN     128
#define BK     32
#define KPAD   40          // 32 + 8 pad (80B row stride, 16B aligned)
#define MINNORM 1e-15f

// ---------------------------------------------------------------------------
// Block reduction (256 threads)
// ---------------------------------------------------------------------------
__device__ __forceinline__ float block_reduce_sum_256(float v, float* red) {
  const int tid = threadIdx.x;
  red[tid] = v;
  __syncthreads();
#pragma unroll
  for (int s = 128; s > 0; s >>= 1) {
    if (tid < s) red[tid] += red[tid + s];
    __syncthreads();
  }
  float r = red[0];
  __syncthreads();
  return r;
}

// ---------------------------------------------------------------------------
// Kernel 1: per-row safe norm of x
// ---------------------------------------------------------------------------
__global__ __launch_bounds__(256) void row_norm_k(const float* __restrict__ X,
                                                  float* __restrict__ xn) {
  __shared__ float red[256];
  const int row = blockIdx.x;
  const float* p = X + (size_t)row * DIN;
  float s = 0.f;
#pragma unroll
  for (int i = 0; i < DIN / 256; ++i) {
    float v = p[threadIdx.x + i * 256];
    s += v * v;
  }
  float t = block_reduce_sum_256(s, red);
  if (threadIdx.x == 0) xn[row] = sqrtf(fmaxf(t, MINNORM));
}

// ---------------------------------------------------------------------------
// Kernel 2: hyp_bias = project(expmap0(bias)), plus |hyp_bias|^2
// ---------------------------------------------------------------------------
__global__ __launch_bounds__(256) void bias_k(const float* __restrict__ b,
                                              float* __restrict__ hb,
                                              float* __restrict__ hbn2) {
  __shared__ float red[256];
  const int tid = threadIdx.x;
  float vloc[4];
  float s = 0.f;
#pragma unroll
  for (int i = 0; i < 4; ++i) {
    vloc[i] = b[tid + i * 256];
    s += vloc[i] * vloc[i];
  }
  float un = sqrtf(fmaxf(block_reduce_sum_256(s, red), MINNORM));
  float sc = tanhf(un) / un;                 // sqc == 1
  float h[4];
  float s2 = 0.f;
#pragma unroll
  for (int i = 0; i < 4; ++i) { h[i] = sc * vloc[i]; s2 += h[i] * h[i]; }
  float hn = sqrtf(fmaxf(block_reduce_sum_256(s2, red), MINNORM));
  const float maxn = 1.0f - 4e-3f;
  float pf = (hn > maxn) ? (maxn / hn) : 1.0f;
  float y2acc = 0.f;
#pragma unroll
  for (int i = 0; i < 4; ++i) {
    h[i] *= pf;
    hb[tid + i * 256] = h[i];
    y2acc += h[i] * h[i];
  }
  float y2 = block_reduce_sum_256(y2acc, red);
  if (tid == 0) hbn2[0] = y2;
}

// ---------------------------------------------------------------------------
// Kernel 3: GEMM  OUT[n,o] = sum_k X[n,k] * W[o,k]   (bf16x3 WMMA)
// 256 threads = 8 waves. Waves laid out 4(M) x 2(N); each wave: 32x64 tile
// = 2x4 WMMA accumulators of 16x16.
// ---------------------------------------------------------------------------
__global__ __launch_bounds__(256) void gemm_bf16x3(const float* __restrict__ X,
                                                   const float* __restrict__ W,
                                                   float* __restrict__ OUT) {
  __shared__ __align__(16) __bf16 Ah[2][BM][KPAD];
  __shared__ __align__(16) __bf16 Al[2][BM][KPAD];
  __shared__ __align__(16) __bf16 Bh[2][BN][KPAD];
  __shared__ __align__(16) __bf16 Bl[2][BN][KPAD];

  const int tid  = threadIdx.x;
  const int lane = tid & 31;
  const int wid  = tid >> 5;
  const int wm   = wid & 3;     // 4 waves in M
  const int wn   = wid >> 2;    // 2 waves in N
  const int half = lane >> 4;   // lane half (ISA 16-bit layouts)
  const int l16  = lane & 15;

  const int bm = blockIdx.y * BM;
  const int bn = blockIdx.x * BN;

  v8f acc[2][4];
#pragma unroll
  for (int i = 0; i < 2; ++i)
#pragma unroll
    for (int j = 0; j < 4; ++j)
      acc[i][j] = (v8f){0.f, 0.f, 0.f, 0.f, 0.f, 0.f, 0.f, 0.f};

  float4 areg[4], breg[4];

  // Stage global fp32 tiles into registers (128x32 each for A and B).
  auto gload = [&](int kt) {
    const int k0 = kt * BK;
#pragma unroll
    for (int it = 0; it < 4; ++it) {
      int idx = it * 256 + tid;
      int r = idx >> 3;
      int c = (idx & 7) << 2;
      areg[it] = *(const float4*)(X + (size_t)(bm + r) * DIN + k0 + c);
      breg[it] = *(const float4*)(W + (size_t)(bn + r) * DIN + k0 + c);
    }
  };

  // Convert fp32 -> (bf16 hi, bf16 lo) and store into LDS tiles.
  auto lstore = [&](int buf) {
#pragma unroll
    for (int it = 0; it < 4; ++it) {
      int idx = it * 256 + tid;
      int r = idx >> 3;
      int c = (idx & 7) << 2;
      float av[4] = {areg[it].x, areg[it].y, areg[it].z, areg[it].w};
      float bv[4] = {breg[it].x, breg[it].y, breg[it].z, breg[it].w};
#pragma unroll
      for (int j = 0; j < 4; ++j) {
        __bf16 ah = (__bf16)av[j];
        __bf16 al = (__bf16)(av[j] - (float)ah);
        Ah[buf][r][c + j] = ah;
        Al[buf][r][c + j] = al;
        __bf16 bh = (__bf16)bv[j];
        __bf16 bl = (__bf16)(bv[j] - (float)bh);
        Bh[buf][r][c + j] = bh;
        Bl[buf][r][c + j] = bl;
      }
    }
  };

  // A fragment (16x32, M x K): lane m=l16 holds row m; lanes 0-15 carry
  // K in {0..7, 16..23}, lanes 16-31 carry K in {8..15, 24..31}.
  auto ldfragA = [&](const __bf16* rowp) -> v16bf {
    v8bf x0 = *(const v8bf*)(rowp + half * 8);
    v8bf x1 = *(const v8bf*)(rowp + 16 + half * 8);
    v16bf r;
#pragma unroll
    for (int i = 0; i < 8; ++i) { r[i] = x0[i]; r[i + 8] = x1[i]; }
    return r;
  };
  // B fragment (32x16, K x N): lane n=l16 holds column n; lanes 0-15 carry
  // K in {0..7, 16..23}, lanes 16-31 carry K in {8..15, 24..31}.
  auto ldfragB = [&](const __bf16* rowp) -> v16bf {
    v8bf x0 = *(const v8bf*)(rowp + half * 8);
    v8bf x1 = *(const v8bf*)(rowp + 16 + half * 8);
    v16bf r;
#pragma unroll
    for (int i = 0; i < 8; ++i) { r[i] = x0[i]; r[i + 8] = x1[i]; }
    return r;
  };

  gload(0);
  lstore(0);
  __syncthreads();

  const int KT = DIN / BK;   // 32
  for (int kt = 0; kt < KT; ++kt) {
    const int cur = kt & 1;
    if (kt + 1 < KT) gload(kt + 1);

    v16bf bhf[4], blf[4];
#pragma unroll
    for (int nt = 0; nt < 4; ++nt) {
      bhf[nt] = ldfragB(&Bh[cur][wn * 64 + nt * 16 + l16][0]);
      blf[nt] = ldfragB(&Bl[cur][wn * 64 + nt * 16 + l16][0]);
    }
#pragma unroll
    for (int mt = 0; mt < 2; ++mt) {
      v16bf ah = ldfragA(&Ah[cur][wm * 32 + mt * 16 + l16][0]);
      v16bf al = ldfragA(&Al[cur][wm * 32 + mt * 16 + l16][0]);
#pragma unroll
      for (int nt = 0; nt < 4; ++nt) {
        acc[mt][nt] = __builtin_amdgcn_wmma_f32_16x16x32_bf16(
            false, ah, false, bhf[nt], (short)0, acc[mt][nt], false, false);
        acc[mt][nt] = __builtin_amdgcn_wmma_f32_16x16x32_bf16(
            false, ah, false, blf[nt], (short)0, acc[mt][nt], false, false);
        acc[mt][nt] = __builtin_amdgcn_wmma_f32_16x16x32_bf16(
            false, al, false, bhf[nt], (short)0, acc[mt][nt], false, false);
      }
    }

    if (kt + 1 < KT) lstore(cur ^ 1);
    __syncthreads();
  }

  // C/D layout: VGPR r -> row (half*8 + r), lane l16 -> column.
#pragma unroll
  for (int mt = 0; mt < 2; ++mt)
#pragma unroll
    for (int nt = 0; nt < 4; ++nt)
#pragma unroll
      for (int r = 0; r < 8; ++r) {
        int row = bm + wm * 32 + mt * 16 + half * 8 + r;
        int col = bn + wn * 64 + nt * 16 + l16;
        OUT[(size_t)row * DOUT + col] = acc[mt][nt][r];
      }
}

// ---------------------------------------------------------------------------
// Kernel 4: per-row hyperbolic epilogue (in place on OUT).
// res = tanh(|mx|/|x| * artanh(|x|)) * mx/|mx|; project; mobius_add(res, hb);
// project; c == 1 throughout.
// ---------------------------------------------------------------------------
__global__ __launch_bounds__(256) void final_k(float* __restrict__ OUT,
                                               const float* __restrict__ xn,
                                               const float* __restrict__ hb,
                                               const float* __restrict__ hbn2) {
  __shared__ float red[256];
  const int row = blockIdx.x;
  const int tid = threadIdx.x;
  float* p = OUT + (size_t)row * DOUT;

  float m[4], hbv[4];
  float s = 0.f;
#pragma unroll
  for (int i = 0; i < 4; ++i) {
    int c = tid + i * 256;
    m[i] = p[c];
    hbv[i] = hb[c];
    s += m[i] * m[i];
  }
  float mxn2 = block_reduce_sum_256(s, red);
  float mx_norm = sqrtf(fmaxf(mxn2, MINNORM));
  float x_norm = xn[row];

  float u  = fminf(x_norm, 1.0f - 1e-5f);       // artanh clip (norm >= 0)
  float at = atanhf(u);
  float t  = tanhf(mx_norm / x_norm * at);
  float scale = t / mx_norm;                    // / (mx_norm * sqc)
  if (mxn2 == 0.0f) scale = 0.0f;               // all-zero mx row

  const float maxn = 1.0f - 4e-3f;
  float rn2 = scale * scale * mxn2;
  float rn  = sqrtf(fmaxf(rn2, MINNORM));
  float pf  = (rn > maxn) ? (maxn / rn) : 1.0f;

  float r_[4];
  float xy_acc = 0.f;
#pragma unroll
  for (int i = 0; i < 4; ++i) {
    r_[i] = scale * pf * m[i];
    xy_acc += r_[i] * hbv[i];
  }
  float x2 = (rn > maxn) ? (maxn * maxn) : rn2;
  float xy = block_reduce_sum_256(xy_acc, red);
  float y2 = hbn2[0];

  float A = 1.0f + 2.0f * xy + y2;
  float B = 1.0f - x2;
  float den = fmaxf(1.0f + 2.0f * xy + x2 * y2, MINNORM);

  float o[4];
  float on_acc = 0.f;
#pragma unroll
  for (int i = 0; i < 4; ++i) {
    o[i] = (A * r_[i] + B * hbv[i]) / den;
    on_acc += o[i] * o[i];
  }
  float on  = sqrtf(fmaxf(block_reduce_sum_256(on_acc, red), MINNORM));
  float pf2 = (on > maxn) ? (maxn / on) : 1.0f;
#pragma unroll
  for (int i = 0; i < 4; ++i) p[tid + i * 256] = o[i] * pf2;
}

// ---------------------------------------------------------------------------
extern "C" void kernel_launch(void* const* d_in, const int* in_sizes, int n_in,
                              void* d_out, int out_size, void* d_ws, size_t ws_size,
                              hipStream_t stream) {
  (void)in_sizes; (void)n_in; (void)out_size; (void)ws_size;
  const float* x = (const float*)d_in[0];
  const float* w = (const float*)d_in[1];
  const float* b = (const float*)d_in[2];
  float* out = (float*)d_out;

  float* wsf  = (float*)d_ws;
  float* xn   = wsf;                       // N_ROWS floats
  float* hb   = wsf + N_ROWS;              // DOUT floats
  float* hbn2 = wsf + N_ROWS + DOUT;       // 1 float

  row_norm_k<<<N_ROWS, 256, 0, stream>>>(x, xn);
  bias_k<<<1, 256, 0, stream>>>(b, hb, hbn2);

  dim3 grid(DOUT / BN, N_ROWS / BM);
  gemm_bf16x3<<<grid, 256, 0, stream>>>(x, w, out);

  final_k<<<N_ROWS, 256, 0, stream>>>(out, xn, hb, hbn2);
}
